// LSTM_LM_74071005987568
// MI455X (gfx1250) — compile-verified, matrix operand order checked by reading, not run on hardware
//
#include <hip/hip_runtime.h>
#include <hip/hip_bf16.h>

// Mogrifier BiLSTM LM encoder for MI455X (gfx1250).
// - bf16 WMMA (v_wmma_f32_16x16x32_bf16) for every matmul, fp32 accumulate.
// - Weights converted fp32->bf16 once per direction-layer into a ~134MB
//   staging region that stays resident in the 192MB L2 across the 32
//   sequential timesteps (workload is weight-bandwidth bound, M=8 batch).
// - Batch padded 8->16 so every GEMM is exact 16x16x32 WMMA tiles.
// - Gates GEMM fused with the LSTM pointwise update (h_bf double-buffered
//   to avoid the intra-kernel RAW race on the Whh operand).
// - Dual accumulators per wave to keep two WMMA/load chains in flight.
// Requires ws_size >= ~153 MB.

#define VOCAB 12000
#define EMB   128
#define FACT  300
#define HID   2048
#define GATE  (4*HID)      // 8192
#define SEQ   32
#define BATCH 8
#define MPAD  16
#define IN0P  320          // FACT padded to multiple of 32
#define INL   4096         // 2*HID

typedef __attribute__((ext_vector_type(16))) __bf16 bf16x16;
typedef __attribute__((ext_vector_type(8)))  float  f32x8;

__device__ __forceinline__ float sigf(float z) {
    return 1.0f / (1.0f + __expf(-z));
}

// One 16x16x32 WMMA step. A: activations [16 x K] row-major bf16 (ld = ldA).
// Wt: weight tile base = W + ntile*ldW, rows are output features, K contiguous.
__device__ __forceinline__ f32x8 wmma_step(const __bf16* __restrict__ A, int ldA,
                                           const __bf16* __restrict__ Wt, int ldW,
                                           int kbase, f32x8 acc) {
    const int lane = threadIdx.x & 31;
    const int idx  = lane & 15;
    const int half = lane >> 4;
    bf16x16 a = *(const bf16x16*)(A  + (size_t)idx * ldA + kbase + half * 16);
    bf16x16 b = *(const bf16x16*)(Wt + (size_t)idx * ldW + kbase + half * 16);
    return __builtin_amdgcn_wmma_f32_16x16x32_bf16(false, a, false, b,
                                                   (short)0, acc, false, false);
}

// ---------------- embedding + factorization ----------------
// xseq[t][m][k] = (m<8 && k<300) ? emb[tok(m,t)] . emb_w[k] + emb_b[k] : 0
__global__ void k_embed(const int* __restrict__ tokens,
                        const float* __restrict__ emb,
                        const float* __restrict__ ew,
                        const float* __restrict__ eb,
                        float* __restrict__ xseq) {
    int idx = blockIdx.x * 256 + threadIdx.x;
    if (idx >= SEQ * MPAD * IN0P) return;
    int k = idx % IN0P;
    int m = (idx / IN0P) % MPAD;
    int t = idx / (IN0P * MPAD);
    float v = 0.0f;
    if (m < BATCH && k < FACT) {
        int tok = tokens[m * SEQ + t];
        const float* e = emb + (size_t)tok * EMB;
        const float* w = ew + (size_t)k * EMB;
        float s = 0.0f;
        #pragma unroll 8
        for (int j = 0; j < EMB; ++j) s += e[j] * w[j];
        v = s + eb[k];
    }
    xseq[idx] = v;
}

// ---------------- fp32 -> bf16 weight staging (with zero pad) -------------
__global__ void k_convert(const float* __restrict__ src, int Nsrc, int Ksrc,
                          __bf16* __restrict__ dst, int Kdst, long long total) {
    long long idx = (long long)blockIdx.x * 256 + threadIdx.x;
    if (idx >= total) return;
    int n = (int)(idx / Kdst);
    int k = (int)(idx % Kdst);
    float v = (n < Nsrc && k < Ksrc) ? src[(size_t)n * Ksrc + k] : 0.0f;
    dst[idx] = (__bf16)v;
}

// ---------------- zero h / c / both h_bf16 buffers ----------------
__global__ void k_zero_state(float* __restrict__ h, __bf16* __restrict__ hb0,
                             __bf16* __restrict__ hb1, float* __restrict__ c) {
    int i = blockIdx.x * 256 + threadIdx.x;
    if (i < MPAD * HID) {
        h[i] = 0.0f; c[i] = 0.0f;
        hb0[i] = (__bf16)0.0f; hb1[i] = (__bf16)0.0f;
    }
}

// ---------------- mogrifier GEMM: out = 2*sigmoid(A @ W^T) * src ----------
// One workgroup (4 waves) per 16-wide N tile; waves split K, LDS reduce.
__global__ void k_gemm_mog(const __bf16* __restrict__ A, int ldA, int Ktot,
                           const __bf16* __restrict__ W, int ldW,
                           const float* __restrict__ src,
                           float* __restrict__ outF, __bf16* __restrict__ outB,
                           int ldN) {
    __shared__ float red[4][256];
    const int ntile = blockIdx.x * 16;
    const int wave  = threadIdx.x >> 5;
    const int lane  = threadIdx.x & 31;
    const __bf16* Wt = W + (size_t)ntile * ldW;
    f32x8 acc0 = {}, acc1 = {};
    const int nks = Ktot >> 5;
    int ks = wave;
    for (; ks + 4 < nks; ks += 8) {
        acc0 = wmma_step(A, ldA, Wt, ldW, ks * 32, acc0);
        acc1 = wmma_step(A, ldA, Wt, ldW, (ks + 4) * 32, acc1);
    }
    if (ks < nks) acc0 = wmma_step(A, ldA, Wt, ldW, ks * 32, acc0);
    if (wave) {
        #pragma unroll
        for (int i = 0; i < 8; ++i) red[wave][lane * 8 + i] = acc0[i] + acc1[i];
    }
    __syncthreads();
    if (wave == 0) {
        #pragma unroll
        for (int i = 0; i < 8; ++i) {
            float v = acc0[i] + acc1[i] + red[1][lane * 8 + i]
                    + red[2][lane * 8 + i] + red[3][lane * 8 + i];
            int m = ((lane >> 4) << 3) + i;        // ISA C layout
            int n = ntile + (lane & 15);
            float r = 2.0f * sigf(v) * src[(size_t)m * ldN + n];
            outF[(size_t)m * ldN + n] = r;
            outB[(size_t)m * ldN + n] = (__bf16)r;
        }
    }
}

// ------ fused gates GEMM + LSTM update ------------------------------------
// Grid: HID/16 workgroups, 512 threads = 16 waves = 4 gates x 4 K-splits.
// gates(g,n) = x@Wih^T + h@Whh^T + bih + bhh for rows g*HID+ntile..+16.
// Wave 0 applies the LSTM pointwise update and writes h (fp32 + bf16-next),
// c, and the output sequence. H (bf16) is read-only here; new h goes to
// hBnext (double buffer) to avoid an intra-kernel race.
__global__ void k_gates_lstm(const __bf16* __restrict__ X, int K1,
                             const __bf16* __restrict__ Wih,
                             const __bf16* __restrict__ H,
                             const __bf16* __restrict__ Whh,
                             const float* __restrict__ bih,
                             const float* __restrict__ bhh,
                             float* __restrict__ hF,
                             __bf16* __restrict__ hBnext,
                             float* __restrict__ cF,
                             float* __restrict__ yseq,
                             int t, int dir) {
    __shared__ float red[16][256];
    const int ntile = blockIdx.x * 16;          // n within HID
    const int wid   = threadIdx.x >> 5;         // 0..15
    const int lane  = threadIdx.x & 31;
    const int g     = wid >> 2;                 // gate index: i,f,g,o
    const int ksub  = wid & 3;                  // K split
    const int nrow  = g * HID + ntile;          // row block in GATE dim
    f32x8 acc0 = {}, acc1 = {};
    {
        const __bf16* W1 = Wih + (size_t)nrow * K1;
        const int nks = K1 >> 5;
        int ks = ksub;
        for (; ks + 4 < nks; ks += 8) {
            acc0 = wmma_step(X, K1, W1, K1, ks * 32, acc0);
            acc1 = wmma_step(X, K1, W1, K1, (ks + 4) * 32, acc1);
        }
        if (ks < nks) acc0 = wmma_step(X, K1, W1, K1, ks * 32, acc0);
    }
    {
        const __bf16* W2 = Whh + (size_t)nrow * HID;
        const int nks = HID >> 5;
        int ks = ksub;
        for (; ks + 4 < nks; ks += 8) {
            acc0 = wmma_step(H, HID, W2, HID, ks * 32, acc0);
            acc1 = wmma_step(H, HID, W2, HID, (ks + 4) * 32, acc1);
        }
        if (ks < nks) acc0 = wmma_step(H, HID, W2, HID, ks * 32, acc0);
    }
    #pragma unroll
    for (int i = 0; i < 8; ++i) red[wid][lane * 8 + i] = acc0[i] + acc1[i];
    __syncthreads();
    const int n = ntile + (lane & 15);
    if (wid < 4) {                              // reduce K-splits per gate
        float b = bih[wid * HID + n] + bhh[wid * HID + n];
        #pragma unroll
        for (int i = 0; i < 8; ++i) {
            int j = lane * 8 + i;
            red[wid * 4][j] = red[wid * 4][j] + red[wid * 4 + 1][j]
                            + red[wid * 4 + 2][j] + red[wid * 4 + 3][j] + b;
        }
    }
    __syncthreads();
    if (wid == 0) {                             // LSTM pointwise update
        #pragma unroll
        for (int i = 0; i < 8; ++i) {
            int j = lane * 8 + i;
            int m = ((lane >> 4) << 3) + i;
            float ig = sigf(red[0][j]);
            float fg = sigf(red[4][j]);
            float gg = tanhf(red[8][j]);
            float og = sigf(red[12][j]);
            size_t hi = (size_t)m * HID + n;
            float cn = fg * cF[hi] + ig * gg;
            float hn = og * tanhf(cn);
            cF[hi] = cn;
            hF[hi] = hn;
            hBnext[hi] = (__bf16)hn;
            yseq[((size_t)t * MPAD + m) * INL + dir * HID + n] = hn;
        }
    }
}

// ---------------- final transpose [S][16][4096] -> [B][S][4096] ----------
__global__ void k_output(const float* __restrict__ yseq, float* __restrict__ out) {
    int idx = blockIdx.x * 256 + threadIdx.x;
    if (idx >= BATCH * SEQ * INL) return;
    int j = idx & (INL - 1);
    int s = (idx >> 12) & (SEQ - 1);
    int b = idx >> 17;
    out[idx] = yseq[((size_t)s * MPAD + b) * INL + j];
}

extern "C" void kernel_launch(void* const* d_in, const int* in_sizes, int n_in,
                              void* d_out, int out_size, void* d_ws, size_t ws_size,
                              hipStream_t stream) {
    (void)in_sizes; (void)n_in; (void)out_size; (void)ws_size;
    const int*   tokens    = (const int*)d_in[0];
    const float* embedding = (const float*)d_in[1];
    const float* emb_w     = (const float*)d_in[2];
    const float* emb_b     = (const float*)d_in[3];
    // rnn_params flattened: layer-major, dir-major, keys Wih,Whh,bih,bhh,Q,R
    auto P = [&](int l, int d, int which) -> const float* {
        return (const float*)d_in[4 + (l * 2 + d) * 6 + which];
    };

    // ---- carve workspace (all regions written before read) ----
    char* p = (char*)d_ws;
    auto carve = [&](size_t bytes) -> char* {
        char* r = p; p += (bytes + 255) & ~(size_t)255; return r;
    };
    __bf16* WihB = (__bf16*)carve((size_t)GATE * INL * 2);      //  64 MiB
    __bf16* WhhB = (__bf16*)carve((size_t)GATE * HID * 2);      //  32 MiB
    __bf16* QB   = (__bf16*)carve((size_t)INL * HID * 2);       //  16 MiB
    __bf16* RB   = (__bf16*)carve((size_t)HID * INL * 2);       //  16 MiB
    float*  seqA = (float*)carve((size_t)SEQ * MPAD * INL * 4); //   8 MiB
    float*  seqB = (float*)carve((size_t)SEQ * MPAD * INL * 4); //   8 MiB
    float*  xcur = (float*)carve((size_t)MPAD * INL * 4);
    __bf16* xcurB= (__bf16*)carve((size_t)MPAD * INL * 2);
    float*  hF   = (float*)carve((size_t)MPAD * HID * 4);
    __bf16* hB0  = (__bf16*)carve((size_t)MPAD * HID * 2);
    __bf16* hB1  = (__bf16*)carve((size_t)MPAD * HID * 2);
    float*  cF   = (float*)carve((size_t)MPAD * HID * 4);

    // embedding + factorization -> seqA [S][16][IN0P]
    {
        int total = SEQ * MPAD * IN0P;
        k_embed<<<(total + 255) / 256, 256, 0, stream>>>(tokens, embedding,
                                                         emb_w, emb_b, seqA);
    }

    float* seqIn = seqA;
    float* seqOut = seqB;

    for (int l = 0; l < 3; ++l) {
        const int IN    = (l == 0) ? IN0P : INL;   // padded K/N of layer input
        const int INsrc = (l == 0) ? FACT : INL;   // true size
        for (int d = 0; d < 2; ++d) {
            const float* pWih = P(l, d, 0);
            const float* pWhh = P(l, d, 1);
            const float* pbih = P(l, d, 2);
            const float* pbhh = P(l, d, 3);
            const float* pQ   = P(l, d, 4);
            const float* pR   = P(l, d, 5);

            // stage bf16 weights for this direction-layer (L2-resident after)
            {
                long long tot;
                tot = (long long)GATE * IN;
                k_convert<<<(unsigned)((tot + 255) / 256), 256, 0, stream>>>(
                    pWih, GATE, INsrc, WihB, IN, tot);
                tot = (long long)GATE * HID;
                k_convert<<<(unsigned)((tot + 255) / 256), 256, 0, stream>>>(
                    pWhh, GATE, HID, WhhB, HID, tot);
                tot = (long long)IN * HID;            // Q: rows = N = in
                k_convert<<<(unsigned)((tot + 255) / 256), 256, 0, stream>>>(
                    pQ, INsrc, HID, QB, HID, tot);
                tot = (long long)HID * IN;            // R: rows = N = HID
                k_convert<<<(unsigned)((tot + 255) / 256), 256, 0, stream>>>(
                    pR, HID, INsrc, RB, IN, tot);
            }
            k_zero_state<<<(MPAD * HID + 255) / 256, 256, 0, stream>>>(
                hF, hB0, hB1, cF);

            __bf16* hbCur = hB0;
            __bf16* hbNxt = hB1;
            for (int step = 0; step < SEQ; ++step) {
                const int t = d ? (SEQ - 1 - step) : step;
                const float* xin = seqIn + (size_t)t * MPAD * IN;

                // 5 mogrifier rounds (even: gate x by h; odd: gate h by x)
                for (int i = 0; i < 5; ++i) {
                    if ((i & 1) == 0) {
                        k_gemm_mog<<<IN / 16, 128, 0, stream>>>(
                            hbCur, HID, HID, QB, HID,
                            (i == 0) ? xin : xcur, xcur, xcurB, IN);
                    } else {
                        k_gemm_mog<<<HID / 16, 128, 0, stream>>>(
                            xcurB, IN, IN, RB, IN,
                            hF, hF, hbCur, HID);
                    }
                }
                // fused gates GEMM + LSTM update; writes next-step h into
                // the other bf16 buffer
                k_gates_lstm<<<HID / 16, 512, 0, stream>>>(
                    xcurB, IN, WihB, hbCur, WhhB, pbih, pbhh,
                    hF, hbNxt, cF, seqOut, t, d);
                __bf16* tmp = hbCur; hbCur = hbNxt; hbNxt = tmp;
            }
        }
        float* tmp = seqIn; seqIn = seqOut; seqOut = tmp;
    }

    // after 3 swaps, seqIn holds the last layer's output
    {
        int total = BATCH * SEQ * INL;
        k_output<<<(total + 255) / 256, 256, 0, stream>>>(seqIn, (float*)d_out);
    }
}
